// SepLinearTaylorAttention_25984552141260
// MI455X (gfx1250) — matrix operand order repaired and checked by reading, not compiled, not run on hardware
//
#include <hip/hip_runtime.h>

// ---------- types ----------
typedef _Float16 h2    __attribute__((ext_vector_type(2)));
typedef _Float16 v16h  __attribute__((ext_vector_type(16)));
typedef float    v8f   __attribute__((ext_vector_type(8)));
typedef unsigned int u32x4 __attribute__((ext_vector_type(4)));
typedef int      i32x4 __attribute__((ext_vector_type(4)));
typedef int      i32x8 __attribute__((ext_vector_type(8)));

union AFrag { v16h v; h2 p[8]; };

static __device__ inline h2 mk2(_Float16 a, _Float16 b) { h2 r; r[0] = a; r[1] = b; return r; }

#define WMMA(A, B, C) __builtin_amdgcn_wmma_f32_16x16x32_f16(false, (A), false, (B), (short)0, (C), false, false)

// ---------- problem constants ----------
constexpr int BH = 32;       // B*H = 4*8
constexpr int N  = 4096;
constexpr int D  = 32;
constexpr int FT = 1056;     // 1024 quadratic features + 32 kv1 rows
constexpr int FP = FT / 2;   // 528 K-pairs
constexpr int NS = 4;        // K-dim splits in phase 1

// TDM availability (ROCm7.2 clang-22: 5-arg; amdgpu-toolchain clang-23: 6-arg)
#if defined(__has_builtin)
#if __has_builtin(__builtin_amdgcn_tensor_load_to_lds) && __has_builtin(__builtin_amdgcn_s_wait_tensorcnt)
#define USE_TDM 1
#endif
#endif

// =====================================================================
// Phase 1: kv2[f=32e+d][c] = sum_n k[n,d]k[n,e]v[n,c]   (f16 WMMA, f32 acc)
//          rows 1024..1055 hold kv1[d][c] = sum_n k[n,d]v[n,c]
// LDS: k,v staged f16, transposed + K-pair packed: kst[d][jp] = {k[2jp,d],k[2jp+1,d]}
// =====================================================================
__global__ __launch_bounds__(512) void k_phase1(const float* __restrict__ qkv,
                                                float* __restrict__ part,
                                                float* __restrict__ ksum_p,
                                                float* __restrict__ vsum_p) {
    __shared__ h2 kst[32 * 18];  // 32 d-rows x 16 j-pairs (+2 pad)
    __shared__ h2 vst[32 * 18];

    const int blk = blockIdx.x;
    const int bh = blk / NS, ns = blk % NS;
    const float* kg = qkv + (size_t)(BH + bh) * (N * D);
    const float* vg = qkv + (size_t)(2 * BH + bh) * (N * D);

    const int tid = threadIdx.x;
    const int w = tid >> 5, lane = tid & 31, hi = lane >> 4, m = lane & 15;
    const int e0 = 2 * w;  // this wave owns e = e0, e0+1

    v8f C[2][2][2] = {};   // [e_loc][dhalf][coltile]
    v8f C1 = {};           // kv1 tile for waves 0..3
    float ksacc = 0.f, vsacc = 0.f;

    // A-frag K-pair index per VGPR v (contiguous k order: k = 16*hi + s), in j-pair units
    int jpA[8];
#pragma unroll
    for (int v = 0; v < 8; ++v) jpA[v] = 8 * hi + v;

    // kv1 tile rows for waves 0..3 (direct LDS addressing: no frag selection)
    const int rA1 = ((w & 2) ? 16 : 0) + m;
    const int rB1 = ((w & 1) ? 16 : 0) + m;

    const int sd = tid & 31, sjp = tid >> 5;  // staging coords (512 threads = 32 d x 16 jp)

    for (int c0 = 0; c0 < 1024; c0 += 32) {
        const int n0 = ns * 1024 + c0;
        __syncthreads();
        {
            const size_t g0 = (size_t)(n0 + 2 * sjp) * D + sd;
            kst[sd * 18 + sjp] = mk2((_Float16)kg[g0], (_Float16)kg[g0 + D]);
            vst[sd * 18 + sjp] = mk2((_Float16)vg[g0], (_Float16)vg[g0 + D]);
        }
        __syncthreads();

        AFrag a0, a1, b0, b1, ke[2];
#pragma unroll
        for (int v = 0; v < 8; ++v) {
            const int jp = jpA[v];
            a0.p[v]    = kst[m * 18 + jp];
            a1.p[v]    = kst[(16 + m) * 18 + jp];
            ke[0].p[v] = kst[e0 * 18 + jp];
            ke[1].p[v] = kst[(e0 + 1) * 18 + jp];
            const int jb = 8 * hi + v;  // B pair index (contiguous k order, same as A)
            b0.p[v]    = vst[m * 18 + jb];
            b1.p[v]    = vst[(16 + m) * 18 + jb];
        }

        // all A-temps into distinct registers first (no WMMA->VALU WAR hazard),
        // then the 8 WMMAs back-to-back
        AFrag t[2][2];
#pragma unroll
        for (int el = 0; el < 2; ++el)
#pragma unroll
            for (int v = 0; v < 8; ++v) {
                t[el][0].p[v] = a0.p[v] * ke[el].p[v];
                t[el][1].p[v] = a1.p[v] * ke[el].p[v];
            }
#pragma unroll
        for (int el = 0; el < 2; ++el) {
            C[el][0][0] = WMMA(t[el][0].v, b0.v, C[el][0][0]);
            C[el][0][1] = WMMA(t[el][0].v, b1.v, C[el][0][1]);
            C[el][1][0] = WMMA(t[el][1].v, b0.v, C[el][1][0]);
            C[el][1][1] = WMMA(t[el][1].v, b1.v, C[el][1][1]);
        }
        if (w < 4) {  // kv1 = k^T v : waves 0..3 take (dhalf, coltile) = (w>>1, w&1)
            AFrag ad, bd;
#pragma unroll
            for (int v = 0; v < 8; ++v) {
                ad.p[v] = kst[rA1 * 18 + jpA[v]];
                bd.p[v] = vst[rB1 * 18 + 8 * hi + v];
            }
            C1 = WMMA(ad.v, bd.v, C1);
        }
        if (w == 4) {
#pragma unroll
            for (int jp = 0; jp < 16; ++jp) {
                h2 t2 = kst[lane * 18 + jp];
                ksacc += (float)t2[0] + (float)t2[1];
            }
        }
        if (w == 5) {
#pragma unroll
            for (int jp = 0; jp < 16; ++jp) {
                h2 t2 = vst[lane * 18 + jp];
                vsacc += (float)t2[0] + (float)t2[1];
            }
        }
    }

    float* pb = part + (size_t)(bh * NS + ns) * (FT * D);
#pragma unroll
    for (int el = 0; el < 2; ++el)
#pragma unroll
        for (int dh = 0; dh < 2; ++dh)
#pragma unroll
            for (int ct = 0; ct < 2; ++ct)
#pragma unroll
                for (int r = 0; r < 8; ++r) {
                    int f = 32 * (e0 + el) + 16 * dh + r + 8 * hi;  // C layout rows r / r+8
                    pb[f * D + 16 * ct + m] = C[el][dh][ct][r];
                }
    if (w < 4) {
        int dh = w >> 1, ct = w & 1;
#pragma unroll
        for (int r = 0; r < 8; ++r) {
            int f = 1024 + 16 * dh + r + 8 * hi;
            pb[f * D + 16 * ct + m] = C1[r];
        }
    }
    if (w == 4) ksum_p[(bh * NS + ns) * 32 + lane] = ksacc;
    if (w == 5) vsum_p[(bh * NS + ns) * 32 + lane] = vsacc;
}

// =====================================================================
// Reduce NS partials; emit kv2 K-pair-interleaved f16: kv2p[p][c] = {f=2p, f=2p+1}
// =====================================================================
__global__ __launch_bounds__(256) void k_reduce(const float* __restrict__ part,
                                                const float* __restrict__ ksum_p,
                                                const float* __restrict__ vsum_p,
                                                h2* __restrict__ kv2p,
                                                _Float16* __restrict__ ksum_h,
                                                float* __restrict__ vsum_f) {
    const int bh = blockIdx.x, tid = threadIdx.x;
    const float* pb = part + (size_t)bh * NS * (FT * D);
    for (int idx = tid; idx < FP * D; idx += 256) {
        const int p = idx >> 5, c = idx & 31;
        float s0 = 0.f, s1 = 0.f;
#pragma unroll
        for (int ns = 0; ns < NS; ++ns) {
            const float* q = pb + (size_t)ns * (FT * D) + (size_t)p * (2 * D) + c;
            s0 += q[0];
            s1 += q[D];
        }
        kv2p[(size_t)bh * (FP * D) + idx] = mk2((_Float16)s0, (_Float16)s1);
    }
    if (tid < 32) {
        float s = 0.f, t = 0.f;
#pragma unroll
        for (int ns = 0; ns < NS; ++ns) {
            s += ksum_p[(bh * NS + ns) * 32 + tid];
            t += vsum_p[(bh * NS + ns) * 32 + tid];
        }
        ksum_h[bh * 32 + tid] = (_Float16)s;
        vsum_f[bh * 32 + tid] = t;
    }
}

// =====================================================================
// Phase 2: y = (0.5*Phi(q)@kv2 + q@kv1 + v_sum) / (0.5 qk^2 + qk + 1)
// LDS kv2 image: rows of 32 h2 padded to 34 (TDM pad: 32 DWORDs + 2 DWORDs)
// =====================================================================
constexpr int KROW = 34;                       // padded LDS row stride (dwords/h2)
constexpr int SM_KV2 = FP * KROW * 4;          // 71808 B
constexpr int SM_QS  = 512 * 34 * 2;           // 34816 B (q rows padded 32->34 halves)
constexpr int SM_TOT = SM_KV2 + SM_QS + 64 + 128;  // 106816 B

__global__ __launch_bounds__(256) void k_phase2(const float* __restrict__ qkv,
                                                const h2* __restrict__ kv2p,
                                                const _Float16* __restrict__ ksum_h,
                                                const float* __restrict__ vsum_f,
                                                float* __restrict__ out) {
    extern __shared__ char smem[];
    h2*       kv2s = (h2*)smem;                       // [FP][34]
    _Float16* qs   = (_Float16*)(smem + SM_KV2);      // [512][34]
    _Float16* kss  = (_Float16*)(smem + SM_KV2 + SM_QS);
    float*    vss  = (float*)(smem + SM_KV2 + SM_QS + 64);

    const int bh = blockIdx.x >> 3, seg = blockIdx.x & 7;
    const int tid = threadIdx.x, w = tid >> 5, lane = tid & 31, hi = lane >> 4, m = lane & 15;
    const int nbase = seg * 512;
    const float* qg = qkv + (size_t)bh * (N * D);

    // ---- fill kv2s (TDM async-tensor DMA with LDS padding when available) ----
#ifdef USE_TDM
    if (w == 0) {
        unsigned long long ga = (unsigned long long)(kv2p + (size_t)bh * (FP * D));
        unsigned int lds_off = (unsigned int)(unsigned long long)kv2s;  // low 32 bits = LDS offset
        u32x4 g0;
        g0[0] = 1u;                                                      // count=1, user D#
        g0[1] = lds_off;                                                 // lds_addr
        g0[2] = (unsigned int)ga;                                        // global_addr[31:0]
        g0[3] = (unsigned int)((ga >> 32) & 0x1FFFFFFull) | (2u << 30);  // addr[56:32] | type=2
        i32x8 g1;
        g1[0] = (int)0x03120000;   // data_size=4B | pad_enable | pad_interval=32dw | pad_amount=2dw
        g1[1] = (int)0x42000000;   // tensor_dim0 = 16896 (low16 << 16)
        g1[2] = 0x00010000;        // tensor_dim0 hi16 = 0 | tensor_dim1 = 1
        g1[3] = (int)0x42000000;   // tile_dim0 = 16896
        g1[4] = 0x00000001;        // tile_dim1 = 1, tile_dim2 = 0
        g1[5] = 16896;             // tensor_dim0_stride
        g1[6] = (int)0x42000000;   // tensor_dim1_stride low16 << 16
        g1[7] = 0;
        i32x4 z4 = {0, 0, 0, 0};
#if __clang_major__ >= 23
        i32x8 z8 = {0, 0, 0, 0, 0, 0, 0, 0};
        __builtin_amdgcn_tensor_load_to_lds(g0, g1, z4, z4, z8, 0);
#else
        __builtin_amdgcn_tensor_load_to_lds(g0, g1, z4, z4, 0);
#endif
        __builtin_amdgcn_s_wait_tensorcnt(0);
    }
#else
    {
        const int* src = (const int*)(kv2p + (size_t)bh * (FP * D));
        int* dst = (int*)smem;
        for (int idx = tid; idx < FP * D; idx += 256) {
            int p = idx >> 5, c = idx & 31;
            dst[p * KROW + c] = src[idx];
        }
    }
#endif
    // ---- stage 512 q rows (f16, padded stride 34) ----
    for (int idx = tid; idx < 512 * 32; idx += 256) {
        int row = idx >> 5, d = idx & 31;
        qs[row * 34 + d] = (_Float16)qg[(size_t)(nbase + row) * D + d];
    }
    if (tid < 32) {
        kss[tid] = ksum_h[bh * 32 + tid];
        vss[tid] = vsum_f[bh * 32 + tid];
    }
    __syncthreads();

    // per-lane A-permutation base (contiguous k order: k = 16*hi + s), per row-tile
    const h2* qs2 = (const h2*)qs;  // row stride 17 h2
    AFrag ap[4];
    int lr[4];
#pragma unroll
    for (int t = 0; t < 4; ++t) {
        lr[t] = t * 128 + w * 16 + m;
#pragma unroll
        for (int v = 0; v < 8; ++v) {
            int pi = 8 * hi + v;
            ap[t].p[v] = qs2[lr[t] * 17 + pi];
        }
    }

    v8f Cq[4][2] = {};
    v8f Ck[4] = {};

    for (int e = 0; e < 32; ++e) {
        AFrag b0, b1;
        const int p0 = e * 16 + 8 * hi;  // K-pair row base for this e-block
#pragma unroll
        for (int v = 0; v < 8; ++v) {
            b0.p[v] = kv2s[(p0 + v) * KROW + m];
            b1.p[v] = kv2s[(p0 + v) * KROW + 16 + m];
        }
        // build all 4 A fragments into distinct registers, then 8 WMMAs
        AFrag a[4];
#pragma unroll
        for (int t = 0; t < 4; ++t) {
            _Float16 qe = qs[lr[t] * 34 + e] * (_Float16)0.5f;  // fold the 0.5 Taylor factor
            h2 q2 = mk2(qe, qe);
#pragma unroll
            for (int v = 0; v < 8; ++v) a[t].p[v] = ap[t].p[v] * q2;
        }
#pragma unroll
        for (int t = 0; t < 4; ++t) {
            Cq[t][0] = WMMA(a[t].v, b0.v, Cq[t][0]);
            Cq[t][1] = WMMA(a[t].v, b1.v, Cq[t][1]);
        }
    }
    // linear term (kv1 pair-rows at p=512) + qk via k_sum-broadcast B
    {
        AFrag b0, b1, bk;
        const int p0 = 512 + 8 * hi;
#pragma unroll
        for (int v = 0; v < 8; ++v) {
            b0.p[v] = kv2s[(p0 + v) * KROW + m];
            b1.p[v] = kv2s[(p0 + v) * KROW + 16 + m];
            int kk = 16 * hi + 2 * v;
            bk.p[v] = mk2(kss[kk], kss[kk + 1]);
        }
#pragma unroll
        for (int t = 0; t < 4; ++t) {
            Cq[t][0] = WMMA(ap[t].v, b0.v, Cq[t][0]);
            Cq[t][1] = WMMA(ap[t].v, b1.v, Cq[t][1]);
            Ck[t]    = WMMA(ap[t].v, bk.v, Ck[t]);
        }
    }
    // epilogue: + v_sum, / (0.5 qk^2 + qk + 1)
    const float vs0 = vss[m], vs1 = vss[16 + m];
#pragma unroll
    for (int t = 0; t < 4; ++t) {
#pragma unroll
        for (int r = 0; r < 8; ++r) {
            int row = nbase + t * 128 + w * 16 + r + 8 * hi;  // C layout rows
            float qk = Ck[t][r];
            float nrm = fmaf(qk, fmaf(0.5f, qk, 1.0f), 1.0f);
            float inv = 1.0f / nrm;
            out[(size_t)(bh * N + row) * D + m]      = (Cq[t][0][r] + vs0) * inv;
            out[(size_t)(bh * N + row) * D + 16 + m] = (Cq[t][1][r] + vs1) * inv;
        }
    }
}

// =====================================================================
extern "C" void kernel_launch(void* const* d_in, const int* in_sizes, int n_in,
                              void* d_out, int out_size, void* d_ws, size_t ws_size,
                              hipStream_t stream) {
    (void)in_sizes; (void)n_in; (void)out_size; (void)ws_size;
    const float* qkv = (const float*)d_in[0];
    float* out = (float*)d_out;

    // workspace carve-up (~19.5 MB)
    float* part   = (float*)d_ws;                          // BH*NS*FT*D f32
    float* ksum_p = part + (size_t)BH * NS * FT * D;       // BH*NS*32
    float* vsum_p = ksum_p + (size_t)BH * NS * 32;         // BH*NS*32
    h2* kv2p      = (h2*)(vsum_p + (size_t)BH * NS * 32);  // BH*FP*D h2 (pair-interleaved)
    _Float16* ksum_h = (_Float16*)(kv2p + (size_t)BH * FP * D);  // BH*32 f16
    float* vsum_f    = (float*)(ksum_h + (size_t)BH * 32);       // BH*32 f32

    k_phase1<<<BH * NS, 512, 0, stream>>>(qkv, part, ksum_p, vsum_p);
    k_reduce<<<BH, 256, 0, stream>>>(part, ksum_p, vsum_p, kv2p, ksum_h, vsum_f);
    k_phase2<<<BH * 8, 256, SM_TOT, stream>>>(qkv, kv2p, ksum_h, vsum_f, out);
}
